// LPGBlock_9706626089773
// MI455X (gfx1250) — compile-verified, matrix operand order checked by reading, not run on hardware
//
#include <hip/hip_runtime.h>

typedef __attribute__((ext_vector_type(16))) _Float16 v16h;
typedef __attribute__((ext_vector_type(8)))  _Float16 v8h;
typedef __attribute__((ext_vector_type(8)))  float    v8f;

#define NT_PER_BATCH 198        // 6336 / 32 pixels per tile
#define N_TILES      (16 * 198) // 3168
#define HW8          6336       // 44*144 low-res pixels per batch
#define W8C          144
#define OUTW         1152
#define OUTHW        (352 * 1152)

// per-wave LDS buffer strides (chosen so strided reads hit distinct banks)
#define S0 72   // halves stride, layer0 out (64 ch)
#define S1 40   // halves stride, layer1 out (32 ch)
#define S2 20   // f32 stride,    layer2 out (16 ch)

__device__ __forceinline__ float eluf(float x) {
    return x > 0.0f ? x : (__expf(x) - 1.0f);
}

// Pre-swizzled B fragment (16 halves per lane, 32B) from LDS.
__device__ __forceinline__ v16h frag_from_lds(const _Float16* sFrag, int f, int lane) {
    return *(const v16h*)(sFrag + ((f * 32 + lane) << 4));
}

// A fragment from an LDS [pixel][channel] f16 buffer.
// Lane needs K = {off..off+7, off+16..off+23} of its row -> two ds_load_b128.
__device__ __forceinline__ v16h afrag_from_lds(const _Float16* buf, int pixHalfBase, int off) {
    v8h lo = *(const v8h*)(buf + pixHalfBase + off);
    v8h hi = *(const v8h*)(buf + pixHalfBase + off + 16);
    return __builtin_shufflevector(lo, hi, 0,1,2,3,4,5,6,7,8,9,10,11,12,13,14,15);
}

__global__ __launch_bounds__(128) void lpg_fused_kernel(
    const float* __restrict__ x,
    const float* __restrict__ w0, const float* __restrict__ b0,
    const float* __restrict__ w1, const float* __restrict__ b1,
    const float* __restrict__ w2, const float* __restrict__ b2,
    const float* __restrict__ w3, const float* __restrict__ b3,
    const float* __restrict__ w4, const float* __restrict__ b4,
    const float* __restrict__ wc, const float* __restrict__ bc,
    float* __restrict__ out)
{
    // 21 B-fragments (L0: 4k x 4n, L1: 2k x 2n, L2: 1), 32 lanes x 16 halves each
    __shared__ __align__(32) _Float16 sFrag[21 * 32 * 16];   // 21504 B
    __shared__ float sTailW[192];                            //   768 B
    __shared__ __align__(16) _Float16 sBufA[4][32 * S0];     // 18432 B
    __shared__ __align__(16) _Float16 sBufB[4][32 * S1];     // 10240 B
    __shared__ __align__(16) float    sBufC[4][32 * S2];     // 10240 B

    const int tid  = threadIdx.x;
    const int lane = tid & 31;
    const int wv   = tid >> 5;

    // ---- stage pre-swizzled B fragments (ISA 7.12.2 16-bit operand layout) ----
    for (int e = tid; e < 21 * 32; e += 128) {
        const int f = e >> 5;
        const int L = e & 31;
        const float* w; int C, kb, nt;
        if (f < 16)      { w = w0; C = 128; kb = (f >> 2) * 32;        nt = f & 3; }
        else if (f < 20) { w = w1; C = 64;  kb = ((f - 16) >> 1) * 32; nt = (f - 16) & 1; }
        else             { w = w2; C = 32;  kb = 0;                    nt = 0; }
        const int o   = nt * 16 + (L & 15);
        const int klo = (L >> 4) * 8;
        _Float16* dst = sFrag + ((f * 32 + L) << 4);
        #pragma unroll
        for (int h = 0; h < 16; ++h) {
            const int K = kb + klo + (h < 8 ? h : h + 8);
            dst[h] = (_Float16)w[o * C + K];
        }
    }
    // ---- stage tail (VALU) weights: w3|b3|w4|b4|wc|bc ----
    for (int e = tid; e < 187; e += 128) {
        float v;
        if (e < 128)      v = w3[e];
        else if (e < 136) v = b3[e - 128];
        else if (e < 168) v = w4[e - 136];
        else if (e < 172) v = b4[e - 168];
        else if (e < 184) v = wc[e - 172];
        else              v = bc[e - 184];
        sTailW[e] = v;
    }
    __syncthreads();

    // per-lane biases for WMMA layers (C/D layout: channel = ntile*16 + lane&15)
    const int ch = lane & 15;
    const float bias00 = b0[ch], bias01 = b0[16 + ch], bias02 = b0[32 + ch], bias03 = b0[48 + ch];
    const float bias10 = b1[ch], bias11 = b1[16 + ch];
    const float bias2  = b2[ch];

    // Layer1/layer2 B-fragments are loop-invariant and small: hold in VGPRs.
    v16h bfL1[4];
    #pragma unroll
    for (int f = 0; f < 4; ++f) bfL1[f] = frag_from_lds(sFrag, 16 + f, lane);
    v16h bfL2 = frag_from_lds(sFrag, 20, lane);

    _Float16* bufA = sBufA[wv];
    _Float16* bufB = sBufB[wv];
    float*    bufC = sBufC[wv];

    const int gwave  = blockIdx.x * 4 + wv;
    const int nWaves = gridDim.x * 4;
    const int M   = lane & 15;
    const int klo = (lane >> 4) * 8;
    const int hi  = lane >> 4;

    for (int t = gwave; t < N_TILES; t += nWaves) {
        const int bIdx = t / NT_PER_BATCH;
        const int hw0  = (t % NT_PER_BATCH) * 32;
        const float* xb = x + (size_t)bIdx * 128 * HW8 + hw0;

        // ---- prefetch next tile's x (covers all 128 channel cachelines) ----
        {
            const int t2 = t + nWaves;
            if (t2 < N_TILES) {
                const int b2   = t2 / NT_PER_BATCH;
                const int hw02 = (t2 % NT_PER_BATCH) * 32;
                const float* nb = x + (size_t)b2 * 128 * HW8 + hw02;
                #pragma unroll
                for (int j = 0; j < 4; ++j)
                    __builtin_prefetch(nb + (size_t)(lane * 4 + j) * HW8, 0, 3);
            }
        }

        #pragma unroll
        for (int mt = 0; mt < 2; ++mt) {
            const int pix0 = mt * 16;
            const int pix  = pix0 + M;

            // ---- layer0: 128 -> 64, A gathered from global (f32 -> f16) ----
            v16h a0[4];
            #pragma unroll
            for (int ks = 0; ks < 4; ++ks) {
                float tmp[16];
                #pragma unroll
                for (int h = 0; h < 16; ++h) {
                    const int K = ks * 32 + klo + (h < 8 ? h : h + 8);
                    tmp[h] = xb[(size_t)K * HW8 + pix];
                }
                #pragma unroll
                for (int h = 0; h < 16; ++h) a0[ks][h] = (_Float16)tmp[h];
            }
            #pragma unroll
            for (int nt = 0; nt < 4; ++nt) {
                const float bv = (nt == 0) ? bias00 : (nt == 1) ? bias01 : (nt == 2) ? bias02 : bias03;
                // preload the whole B-fragment chain -> one clause, one wait, 4 WMMAs back-to-back
                v16h bf[4];
                #pragma unroll
                for (int ks = 0; ks < 4; ++ks) bf[ks] = frag_from_lds(sFrag, ks * 4 + nt, lane);
                v8f acc = {bv, bv, bv, bv, bv, bv, bv, bv};
                #pragma unroll
                for (int ks = 0; ks < 4; ++ks)
                    acc = __builtin_amdgcn_wmma_f32_16x16x32_f16(false, a0[ks], false, bf[ks],
                                                                 (short)0, acc, false, false);
                #pragma unroll
                for (int r = 0; r < 8; ++r)
                    bufA[(pix0 + r + 8 * hi) * S0 + nt * 16 + ch] = (_Float16)eluf(acc[r]);
            }

            // ---- layer1: 64 -> 32 (A from LDS, B held in VGPRs) ----
            v16h a1[2];
            a1[0] = afrag_from_lds(bufA, pix * S0, 0 + klo);
            a1[1] = afrag_from_lds(bufA, pix * S0, 32 + klo);
            #pragma unroll
            for (int nt = 0; nt < 2; ++nt) {
                const float bv = nt ? bias11 : bias10;
                v8f acc = {bv, bv, bv, bv, bv, bv, bv, bv};
                #pragma unroll
                for (int ks = 0; ks < 2; ++ks)
                    acc = __builtin_amdgcn_wmma_f32_16x16x32_f16(false, a1[ks], false, bfL1[ks * 2 + nt],
                                                                 (short)0, acc, false, false);
                #pragma unroll
                for (int r = 0; r < 8; ++r)
                    bufB[(pix0 + r + 8 * hi) * S1 + nt * 16 + ch] = (_Float16)eluf(acc[r]);
            }

            // ---- layer2: 32 -> 16 (A from LDS, B held in VGPRs), f32 out for tail ----
            {
                v16h a2 = afrag_from_lds(bufB, pix * S1, klo);
                v8f acc = {bias2, bias2, bias2, bias2, bias2, bias2, bias2, bias2};
                acc = __builtin_amdgcn_wmma_f32_16x16x32_f16(false, a2, false, bfL2,
                                                             (short)0, acc, false, false);
                #pragma unroll
                for (int r = 0; r < 8; ++r)
                    bufC[(pix0 + r + 8 * hi) * S2 + ch] = eluf(acc[r]);
            }
        } // mt

        // ---- VALU tail: one pixel per lane (32 pixels per wave-tile) ----
        {
            const int p = lane;
            const float* h2p = bufC + p * S2;
            float h2v[16];
            #pragma unroll
            for (int k = 0; k < 16; ++k) h2v[k] = h2p[k];

            float h3v[8];
            #pragma unroll
            for (int o = 0; o < 8; ++o) {
                float s = sTailW[128 + o];
                #pragma unroll
                for (int k = 0; k < 16; ++k) s += sTailW[o * 16 + k] * h2v[k];
                h3v[o] = eluf(s);
            }
            float h4v[4];
            #pragma unroll
            for (int o = 0; o < 4; ++o) {
                float s = sTailW[168 + o];
                #pragma unroll
                for (int k = 0; k < 8; ++k) s += sTailW[136 + o * 8 + k] * h3v[k];
                h4v[o] = s;                       // no activation on last reduction
            }
            float y[3];
            #pragma unroll
            for (int o = 0; o < 3; ++o) {
                float s = sTailW[184 + o];
                #pragma unroll
                for (int k = 0; k < 4; ++k) s += sTailW[172 + o * 4 + k] * h4v[k];
                y[o] = s;
            }

            const float PIf = 3.1415926535f;
            const float theta = (1.0f / (1.0f + __expf(-y[0]))) * (PIf / 6.0f);
            const float phi   = (1.0f / (1.0f + __expf(-y[1]))) * (PIf * 2.0f);
            const float dist  = (1.0f / (1.0f + __expf(-y[2]))) * 81.0f;
            const float st = __sinf(theta);
            float n0 = st * __cosf(phi);
            float n1 = st * __sinf(phi);
            float n2 = __cosf(theta);
            float nrm = fmaxf(sqrtf(n0 * n0 + n1 * n1 + n2 * n2), 1e-12f);
            const float inv = 1.0f / nrm;
            n0 *= inv; n1 *= inv; n2 *= inv;

            const int hwp = hw0 + p;
            const int h8  = hwp / W8C;
            const int w8  = hwp - h8 * W8C;
            float* op = out + (size_t)bIdx * OUTHW + (size_t)(h8 * 8) * OUTW + w8 * 8;

            #pragma unroll
            for (int dy = 0; dy < 8; ++dy) {
                const float vv   = ((float)dy - 3.5f) * 0.125f;
                const float den0 = n1 * vv + n2;
                float dv[8];
                #pragma unroll
                for (int dx = 0; dx < 8; ++dx) {
                    const float uu = ((float)dx - 3.5f) * 0.125f;
                    dv[dx] = dist / (n0 * uu + den0);
                }
                float4* rowp = (float4*)(op + (size_t)dy * OUTW);
                rowp[0] = make_float4(dv[0], dv[1], dv[2], dv[3]);
                rowp[1] = make_float4(dv[4], dv[5], dv[6], dv[7]);
            }
        }
    } // tiles
}

extern "C" void kernel_launch(void* const* d_in, const int* in_sizes, int n_in,
                              void* d_out, int out_size, void* d_ws, size_t ws_size,
                              hipStream_t stream) {
    (void)in_sizes; (void)n_in; (void)d_ws; (void)ws_size; (void)out_size;
    const float* x  = (const float*)d_in[0];
    const float* w0 = (const float*)d_in[1];
    const float* b0 = (const float*)d_in[2];
    const float* w1 = (const float*)d_in[3];
    const float* b1 = (const float*)d_in[4];
    const float* w2 = (const float*)d_in[5];
    const float* b2 = (const float*)d_in[6];
    const float* w3 = (const float*)d_in[7];
    const float* b3 = (const float*)d_in[8];
    const float* w4 = (const float*)d_in[9];
    const float* b4 = (const float*)d_in[10];
    const float* wc = (const float*)d_in[11];
    const float* bc = (const float*)d_in[12];
    float* out = (float*)d_out;

    // 264 blocks x 4 waves = 1056 waves; 3168 tiles -> exactly 3 tiles per wave
    dim3 grid(264), block(128);
    lpg_fused_kernel<<<grid, block, 0, stream>>>(x, w0, b0, w1, b1, w2, b2,
                                                 w3, b3, w4, b4, wc, bc, out);
}
